// Model_84387517432579
// MI455X (gfx1250) — matrix operand order repaired and checked by reading, not compiled, not run on hardware
//
#include <hip/hip_runtime.h>
#include <hip/hip_bf16.h>
#include <math.h>

// ---------------------------------------------------------------------------
// Graph-LSTM on MI455X (gfx1250): bf16 WMMA (16x16x32) with f32 accumulation.
// All f32->bf16 conversion / padding / B-operand layout shuffling is done in
// one-pass packing kernels so GEMM inner loops are pure b128-load -> v_wmma.
// ---------------------------------------------------------------------------

typedef __attribute__((ext_vector_type(16))) __bf16 v16bf;
typedef __attribute__((ext_vector_type(8)))  __bf16 v8bf;   // 16-byte chunk
typedef __attribute__((ext_vector_type(4)))  __bf16 v4bf;   // 8-byte chunk
typedef __attribute__((ext_vector_type(8)))  float  v8f;

#define DEV __device__ __forceinline__

static constexpr long N_NODES = 50000;
static constexpr long N_EDGES = 800000;
static constexpr int  D_WORD  = 300;
static constexpr int  D_WORDP = 320;   // padded to 10 k-steps of 32
static constexpr int  D_LINK  = 64;
static constexpr int  D_NODE  = 128;
static constexpr int  D_STATE = 128;
static constexpr int  NCT     = 8;     // 128 output cols / 16

DEV __bf16 f2bf(float f) { return (__bf16)f; }
DEV float sigmoidf_(float x) { return 1.0f / (1.0f + __expf(-x)); }

DEV v8f wmma_bf16(v16bf a, v16bf b, v8f c) {
    return __builtin_amdgcn_wmma_f32_16x16x32_bf16(
        false, a, false, b, (short)0, c, false, false);
}

// B-operand packed offset: tile (ks, ct), lane -> 16 bf16 (32 bytes)
DEV size_t bofs(int ks, int ct, int lane) {
    return ((size_t)(ks * NCT + ct) * 32 + lane) * 16;
}

// Build v16bf A operand from two contiguous 16-byte chunks
DEV v16bf make_a(const __bf16* chunk0, const __bf16* chunk1) {
    v8bf lo = *(const v8bf*)chunk0;
    v8bf hi = *(const v8bf*)chunk1;
    return __builtin_shufflevector(lo, hi, 0,1,2,3,4,5,6,7,8,9,10,11,12,13,14,15);
}

// ---------------------------------------------------------------------------
// Packing kernels (linear, branch-free bulk conversions)
// ---------------------------------------------------------------------------

// emb_token f32 [N_TOKEN,300] -> bf16 [N_TOKEN,320] zero-padded
__global__ __launch_bounds__(256) void pack_tok_kernel(
    const float* __restrict__ src, __bf16* __restrict__ dst, long n_rows)
{
    const long idx = (long)blockIdx.x * 256 + threadIdx.x;
    if (idx >= n_rows * D_WORDP) return;
    const long r = idx / D_WORDP;
    const int  k = (int)(idx - r * D_WORDP);
    dst[idx] = f2bf((k < D_WORD) ? src[r * D_WORD + k] : 0.0f);
}

// plain f32 -> bf16 copy (emb_link)
__global__ __launch_bounds__(256) void pack_copy_kernel(
    const float* __restrict__ src, __bf16* __restrict__ dst, long count)
{
    const long idx = (long)blockIdx.x * 256 + threadIdx.x;
    if (idx < count) dst[idx] = f2bf(src[idx]);
}

// weight [K,128] f32 -> packed B-operand layout bf16 (zero-padded past K)
__global__ __launch_bounds__(256) void pack_w_kernel(
    const float* __restrict__ W, int K, int Ksteps, __bf16* __restrict__ out)
{
    const long idx = (long)blockIdx.x * 256 + threadIdx.x;
    if (idx >= (long)Ksteps * NCT * 32 * 16) return;
    const int e    = (int)(idx & 15);
    const int lane = (int)((idx >> 4) & 31);
    const int ct   = (int)((idx >> 9) & 7);
    const int ks   = (int)(idx >> 12);
    const int lm = lane & 15, hs = lane >> 4;
    const int k = ks * 32 + ((e < 8) ? (hs * 8 + e) : (16 + hs * 8 + (e - 8)));
    const int n = ct * 16 + lm;
    out[idx] = f2bf((k < K) ? W[(long)k * 128 + n] : 0.0f);
}

// f32 state segment [N,128] -> bf16 slice of inp [N,512] at seg offset
__global__ __launch_bounds__(256) void pack_seg_kernel(
    const float* __restrict__ src, __bf16* __restrict__ inp, int segoff)
{
    const long idx = (long)blockIdx.x * 256 + threadIdx.x;   // 4 elems/thread
    if (idx >= N_NODES * (D_STATE / 4)) return;
    const long node = idx >> 5;
    const int  d    = (int)(idx & 31) * 4;
    const float4 v = *(const float4*)&src[node * D_STATE + d];
    v4bf o; o[0] = f2bf(v.x); o[1] = f2bf(v.y); o[2] = f2bf(v.z); o[3] = f2bf(v.w);
    *(v4bf*)&inp[node * 512 + segoff + d] = o;
}

// ---------------------------------------------------------------------------
// Kernel 1: node_emb = tanh(emb_token[i_token] @ Wc + bc) -> bf16 [N,128]
// grid = N/16 blocks x 256 threads (8 waves, one col-tile each)
// ---------------------------------------------------------------------------
__global__ __launch_bounds__(256) void compress_kernel(
    const int* __restrict__ i_token, const __bf16* __restrict__ tokpad,
    const __bf16* __restrict__ WcP, const float* __restrict__ bc,
    __bf16* __restrict__ node_emb)
{
    const int lane = threadIdx.x & 31;
    const int ct   = threadIdx.x >> 5;
    const int lm = lane & 15, hs = lane >> 4;
    const long rowBase = (long)blockIdx.x * 16;
    const int  n = ct * 16 + lm;
    const __bf16* arow = tokpad + (long)i_token[rowBase + lm] * D_WORDP;

    v8f acc = {};
    #pragma unroll
    for (int ks = 0; ks < 10; ++ks) {
        const int kb = ks * 32;
        v16bf a = make_a(arow + kb + hs * 8, arow + kb + 16 + hs * 8);
        v16bf b = *(const v16bf*)(WcP + bofs(ks, ct, lane));
        acc = wmma_bf16(a, b, acc);
    }

    const float bias = bc[n];
    #pragma unroll
    for (int i = 0; i < 8; ++i) {
        const long node = rowBase + hs * 8 + i;
        node_emb[node * D_NODE + n] = f2bf(tanhf(acc[i] + bias));
    }
}

// ---------------------------------------------------------------------------
// Kernel 2 (fused): link_x = tanh([emb_link[i_link] | node_emb[i_from]] @ Wl + bl)
// scattered into x_in[i_to] / x_out[i_from] with f32 atomics.
// grid = E/16 blocks x 256 threads
// ---------------------------------------------------------------------------
__global__ __launch_bounds__(256) void link_kernel(
    const int* __restrict__ i_link, const int* __restrict__ i_from,
    const int* __restrict__ i_to,
    const __bf16* __restrict__ linkemb, const __bf16* __restrict__ node_emb,
    const __bf16* __restrict__ WlP, const float* __restrict__ bl,
    float* __restrict__ x_in, float* __restrict__ x_out)
{
    const int lane = threadIdx.x & 31;
    const int ct   = threadIdx.x >> 5;
    const int lm = lane & 15, hs = lane >> 4;
    const long eBase = (long)blockIdx.x * 16;
    const int  n = ct * 16 + lm;
    const long eA = eBase + lm;
    const __bf16* lrow = linkemb  + (long)i_link[eA] * D_LINK;
    const __bf16* nrow = node_emb + (long)i_from[eA] * D_NODE;

    v8f acc = {};
    #pragma unroll
    for (int ks = 0; ks < 6; ++ks) {
        const int kb = ks * 32;
        // concat boundary at 64 is 32-aligned -> compile-time source select
        const __bf16* src = (kb < D_LINK) ? (lrow + kb) : (nrow + (kb - D_LINK));
        v16bf a = make_a(src + hs * 8, src + 16 + hs * 8);
        v16bf b = *(const v16bf*)(WlP + bofs(ks, ct, lane));
        acc = wmma_bf16(a, b, acc);
    }

    const float bias = bl[n];
    #pragma unroll
    for (int i = 0; i < 8; ++i) {
        const long e = eBase + hs * 8 + i;
        const float v = tanhf(acc[i] + bias);
        atomicAdd(&x_in [(long)i_to  [e] * D_STATE + n], v);
        atomicAdd(&x_out[(long)i_from[e] * D_STATE + n], v);
    }
}

// ---------------------------------------------------------------------------
// Kernel 3: h_in[i_to] += h[i_from] ; h_out[i_from] += h[i_to]
// one thread per (edge, 4 dims)
// ---------------------------------------------------------------------------
__global__ __launch_bounds__(256) void scatter_h_kernel(
    const int* __restrict__ i_from, const int* __restrict__ i_to,
    const float* __restrict__ h,
    float* __restrict__ h_in, float* __restrict__ h_out)
{
    const long idx = (long)blockIdx.x * 256 + threadIdx.x;
    if (idx >= N_EDGES * 32) return;
    const long e = idx >> 5;
    const int  d = (int)(idx & 31) * 4;
    const long f = i_from[e], t = i_to[e];
    const float4 hf = *(const float4*)&h[f * D_STATE + d];
    const float4 ht = *(const float4*)&h[t * D_STATE + d];
    atomicAdd(&h_in [t * D_STATE + d    ], hf.x);
    atomicAdd(&h_in [t * D_STATE + d + 1], hf.y);
    atomicAdd(&h_in [t * D_STATE + d + 2], hf.z);
    atomicAdd(&h_in [t * D_STATE + d + 3], hf.w);
    atomicAdd(&h_out[f * D_STATE + d    ], ht.x);
    atomicAdd(&h_out[f * D_STATE + d + 1], ht.y);
    atomicAdd(&h_out[f * D_STATE + d + 2], ht.z);
    atomicAdd(&h_out[f * D_STATE + d + 3], ht.w);
}

// ---------------------------------------------------------------------------
// Kernel 4: 4 gate GEMMs (K=512 over bf16 inp) + LSTM update.
// A loaded once per K-step, 4 WMMAs. grid = N/16 blocks x 256 threads.
// ---------------------------------------------------------------------------
__global__ __launch_bounds__(256) void gates_kernel(
    const __bf16* __restrict__ inp,
    const __bf16* __restrict__ WiP, const float* __restrict__ bi,
    const __bf16* __restrict__ WoP, const float* __restrict__ bo,
    const __bf16* __restrict__ WfP, const float* __restrict__ bf_,
    const __bf16* __restrict__ WuP, const float* __restrict__ bu,
    float* __restrict__ c, float* __restrict__ h_dst)
{
    const int lane = threadIdx.x & 31;
    const int ct   = threadIdx.x >> 5;
    const int lm = lane & 15, hs = lane >> 4;
    const long rowBase = (long)blockIdx.x * 16;
    const int  n = ct * 16 + lm;
    const __bf16* irow = inp + (rowBase + lm) * 512;

    v8f acc0 = {}, acc1 = {}, acc2 = {}, acc3 = {};
    #pragma unroll
    for (int ks = 0; ks < 16; ++ks) {
        const int kb = ks * 32;
        v16bf a = make_a(irow + kb + hs * 8, irow + kb + 16 + hs * 8);
        v16bf b0 = *(const v16bf*)(WiP + bofs(ks, ct, lane));
        acc0 = wmma_bf16(a, b0, acc0);
        v16bf b1 = *(const v16bf*)(WoP + bofs(ks, ct, lane));
        acc1 = wmma_bf16(a, b1, acc1);
        v16bf b2 = *(const v16bf*)(WfP + bofs(ks, ct, lane));
        acc2 = wmma_bf16(a, b2, acc2);
        v16bf b3 = *(const v16bf*)(WuP + bofs(ks, ct, lane));
        acc3 = wmma_bf16(a, b3, acc3);
    }

    const float bI = bi[n], bO = bo[n], bF = bf_[n], bU = bu[n];
    #pragma unroll
    for (int i = 0; i < 8; ++i) {
        const long node = rowBase + hs * 8 + i;
        const long idx  = node * D_STATE + n;
        const float ig = sigmoidf_(acc0[i] + bI);
        const float og = sigmoidf_(acc1[i] + bO);
        const float fg = sigmoidf_(acc2[i] + bF);
        const float ug = tanhf    (acc3[i] + bU);
        const float cn = fg * c[idx] + ig * ug;
        c[idx]     = cn;
        h_dst[idx] = og * tanhf(cn);
    }
}

// ---------------------------------------------------------------------------
extern "C" void kernel_launch(void* const* d_in, const int* in_sizes, int n_in,
                              void* d_out, int out_size, void* d_ws, size_t ws_size,
                              hipStream_t stream)
{
    (void)n_in; (void)out_size; (void)ws_size;

    const int*   i_token   = (const int*)  d_in[0];
    const int*   i_link    = (const int*)  d_in[1];
    const int*   i_from    = (const int*)  d_in[2];
    const int*   i_to      = (const int*)  d_in[3];
    const float* emb_token = (const float*)d_in[4];
    const float* emb_link  = (const float*)d_in[5];
    const float* Wc = (const float*)d_in[6];  const float* bc = (const float*)d_in[7];
    const float* Wl = (const float*)d_in[8];  const float* bl = (const float*)d_in[9];
    const float* Wi = (const float*)d_in[10]; const float* bi = (const float*)d_in[11];
    const float* Wo = (const float*)d_in[12]; const float* bo = (const float*)d_in[13];
    const float* Wf = (const float*)d_in[14]; const float* bfp= (const float*)d_in[15];
    const float* Wu = (const float*)d_in[16]; const float* bu = (const float*)d_in[17];

    const long n_token = in_sizes[4] / D_WORD;          // 50000 token rows
    const size_t stateBytes = (size_t)N_NODES * D_STATE * sizeof(float);

    char* ws = (char*)d_ws;
    size_t off = 0;
    auto wsalloc = [&](size_t bytes) -> void* {
        void* p = ws + off;
        off = (off + bytes + 255) & ~(size_t)255;
        return p;
    };
    __bf16* tokpad   = (__bf16*)wsalloc((size_t)n_token * D_WORDP * sizeof(__bf16));
    __bf16* linkemb  = (__bf16*)wsalloc((size_t)in_sizes[5] * sizeof(__bf16));
    __bf16* node_emb = (__bf16*)wsalloc((size_t)N_NODES * D_NODE * sizeof(__bf16));
    __bf16* inp      = (__bf16*)wsalloc((size_t)N_NODES * 512 * sizeof(__bf16));
    __bf16* WcP = (__bf16*)wsalloc((size_t)10 * NCT * 32 * 16 * sizeof(__bf16));
    __bf16* WlP = (__bf16*)wsalloc((size_t) 6 * NCT * 32 * 16 * sizeof(__bf16));
    __bf16* WiP = (__bf16*)wsalloc((size_t)16 * NCT * 32 * 16 * sizeof(__bf16));
    __bf16* WoP = (__bf16*)wsalloc((size_t)16 * NCT * 32 * 16 * sizeof(__bf16));
    __bf16* WfP = (__bf16*)wsalloc((size_t)16 * NCT * 32 * 16 * sizeof(__bf16));
    __bf16* WuP = (__bf16*)wsalloc((size_t)16 * NCT * 32 * 16 * sizeof(__bf16));
    float*  x_in  = (float*)wsalloc(stateBytes);
    float*  x_out = (float*)wsalloc(stateBytes);
    float*  h_in  = (float*)wsalloc(stateBytes);
    float*  h_out = (float*)wsalloc(stateBytes);
    float*  h     = (float*)wsalloc(stateBytes);
    float*  c     = (float*)wsalloc(stateBytes);

    auto nblk = [](long total) { return (unsigned)((total + 255) / 256); };

    // ---- packing passes ----
    pack_tok_kernel <<<nblk(n_token * D_WORDP), 256, 0, stream>>>(emb_token, tokpad, n_token);
    pack_copy_kernel<<<nblk(in_sizes[5]),       256, 0, stream>>>(emb_link, linkemb, in_sizes[5]);
    pack_w_kernel<<<nblk((long)10 * 4096), 256, 0, stream>>>(Wc, D_WORD, 10, WcP);
    pack_w_kernel<<<nblk((long) 6 * 4096), 256, 0, stream>>>(Wl, 192, 6, WlP);
    pack_w_kernel<<<nblk((long)16 * 4096), 256, 0, stream>>>(Wi, 512, 16, WiP);
    pack_w_kernel<<<nblk((long)16 * 4096), 256, 0, stream>>>(Wo, 512, 16, WoP);
    pack_w_kernel<<<nblk((long)16 * 4096), 256, 0, stream>>>(Wf, 512, 16, WfP);
    pack_w_kernel<<<nblk((long)16 * 4096), 256, 0, stream>>>(Wu, 512, 16, WuP);

    hipMemsetAsync(x_in,  0, stateBytes, stream);
    hipMemsetAsync(x_out, 0, stateBytes, stream);
    hipMemsetAsync(h,     0, stateBytes, stream);
    hipMemsetAsync(c,     0, stateBytes, stream);

    // ---- node features ----
    compress_kernel<<<(unsigned)(N_NODES / 16), 256, 0, stream>>>(
        i_token, tokpad, WcP, bc, node_emb);

    // ---- link features + fused segment-sum scatter ----
    link_kernel<<<(unsigned)(N_EDGES / 16), 256, 0, stream>>>(
        i_link, i_from, i_to, linkemb, node_emb, WlP, bl, x_in, x_out);

    // x halves of inp are layer-invariant
    pack_seg_kernel<<<nblk(N_NODES * 32), 256, 0, stream>>>(x_in,  inp,   0);
    pack_seg_kernel<<<nblk(N_NODES * 32), 256, 0, stream>>>(x_out, inp, 128);

    // ---- 2 LSTM layers ----
    for (int layer = 0; layer < 2; ++layer) {
        hipMemsetAsync(h_in,  0, stateBytes, stream);
        hipMemsetAsync(h_out, 0, stateBytes, stream);
        if (layer > 0) {   // layer 0: h == 0 -> segment sums are zero
            scatter_h_kernel<<<nblk(N_EDGES * 32), 256, 0, stream>>>(
                i_from, i_to, h, h_in, h_out);
        }
        pack_seg_kernel<<<nblk(N_NODES * 32), 256, 0, stream>>>(h_in,  inp, 256);
        pack_seg_kernel<<<nblk(N_NODES * 32), 256, 0, stream>>>(h_out, inp, 384);

        float* h_dst = (layer == 1) ? (float*)d_out : h;
        gates_kernel<<<(unsigned)(N_NODES / 16), 256, 0, stream>>>(
            inp, WiP, bi, WoP, bo, WfP, bfp, WuP, bu, c, h_dst);
    }
}
